// EdgeGAT_738734375627
// MI455X (gfx1250) — matrix operand order repaired and checked by reading, not compiled, not run on hardware
//
#include <hip/hip_runtime.h>

#define HIDC 16

typedef __attribute__((ext_vector_type(16))) __bf16 v16bf;
typedef __attribute__((ext_vector_type(8)))  float  v8f;

union BFVec { v16bf v; unsigned short u[16]; };

__device__ __forceinline__ unsigned short f2bf(float f) {
  unsigned u = __float_as_uint(f);
  u += 0x7FFFu + ((u >> 16) & 1u);      // round-to-nearest-even bf16
  return (unsigned short)(u >> 16);
}

// Monotone order-preserving float<->uint encoding (for atomicMax on floats incl. negatives)
__device__ __forceinline__ unsigned encOrd(float f) {
  unsigned u = __float_as_uint(f);
  return (u & 0x80000000u) ? ~u : (u | 0x80000000u);
}
__device__ __forceinline__ float decOrd(unsigned u) {
  return __uint_as_float((u & 0x80000000u) ? (u & 0x7FFFFFFFu) : ~u);
}
#define ENC_NEG_INF 0x007FFFFFu  // encOrd(-inf)

__device__ __forceinline__ float lrelu(float v) { return v > 0.f ? v : 0.2f * v; }

// ---------------- init kernels (re-run every launch; graph-capture safe) ----------
__global__ void init1_k(float* __restrict__ out1, float* __restrict__ den1,
                        unsigned* __restrict__ m1, int n) {
  unsigned i = blockIdx.x * blockDim.x + threadIdx.x;
  unsigned tot = (unsigned)n * 64u;
  if (i >= tot) return;
  out1[i] = 0.f;
  if (i < (unsigned)n * 4u) { den1[i] = 0.f; m1[i] = ENC_NEG_INF; }
}

__global__ void init2_k(float* __restrict__ out2, float* __restrict__ den2,
                        unsigned* __restrict__ m2, int n) {
  unsigned i = blockIdx.x * blockDim.x + threadIdx.x;
  unsigned tot = (unsigned)n * 16u;
  if (i >= tot) return;
  out2[i] = 0.f;
  if (i < (unsigned)n) { den2[i] = 0.f; m2[i] = ENC_NEG_INF; }
}

// ---------------- GEMM1: h1[N,64] = x[N,32] @ W1[32,64] via bf16 WMMA ------------
__global__ void gemm1_wmma_k(const float* __restrict__ x, const float* __restrict__ W,
                             float* __restrict__ h, int n) {
  int lane = threadIdx.x & 31;
  int wave = blockIdx.x * (blockDim.x >> 5) + (threadIdx.x >> 5);
  int rowBase = wave * 16;
  if (rowBase >= n) return;          // wave-uniform exit: EXEC stays all-ones for WMMA
  int lanelo = lane & 15, hi = lane >> 4;
  const float* xr = x + (size_t)(rowBase + lanelo) * 32;
  BFVec a;
#pragma unroll
  for (int i = 0; i < 8; ++i) {      // A 16x32: lanes 0-15 K 0-7/16-23, lanes 16-31 K 8-15/24-31
    int k = ((i < 4) ? 0 : 16) + hi * 8 + (i & 3) * 2;
    a.u[2 * i]     = f2bf(xr[k]);
    a.u[2 * i + 1] = f2bf(xr[k + 1]);
  }
#pragma unroll
  for (int ct = 0; ct < 4; ++ct) {   // 4 column tiles of 16 (reuse A)
    int colBase = ct * 16;
    BFVec b;
#pragma unroll
    for (int j = 0; j < 8; ++j) {    // B 32x16: lanes 0-15 K 0-7/16-23, lanes 16-31 K 8-15/24-31
      int k = ((j < 4) ? 0 : 16) + hi * 8 + (j & 3) * 2;
      b.u[2 * j]     = f2bf(W[(size_t)k * 64 + colBase + lanelo]);
      b.u[2 * j + 1] = f2bf(W[(size_t)(k + 1) * 64 + colBase + lanelo]);
    }
    v8f c = {};
    c = __builtin_amdgcn_wmma_f32_16x16x32_bf16(false, a.v, false, b.v, (short)0, c,
                                                false, false);
#pragma unroll
    for (int r = 0; r < 8; ++r)      // D: VGPR r -> M = r + 8*hi, N = lanelo
      h[(size_t)(rowBase + r + hi * 8) * 64 + colBase + lanelo] = c[r];
  }
}

// ---------------- GEMM2: h2[N,16] = h1b[N,64] @ W2[64,16] (2 chained WMMAs) ------
__global__ void gemm2_wmma_k(const float* __restrict__ hin, const float* __restrict__ W,
                             float* __restrict__ hout, int n) {
  int lane = threadIdx.x & 31;
  int wave = blockIdx.x * (blockDim.x >> 5) + (threadIdx.x >> 5);
  int rowBase = wave * 16;
  if (rowBase >= n) return;
  int lanelo = lane & 15, hi = lane >> 4;
  v8f c = {};
#pragma unroll
  for (int kc = 0; kc < 2; ++kc) {
    const float* xr = hin + (size_t)(rowBase + lanelo) * 64 + kc * 32;
    BFVec a, b;
#pragma unroll
    for (int i = 0; i < 8; ++i) {
      int k = ((i < 4) ? 0 : 16) + hi * 8 + (i & 3) * 2;
      a.u[2 * i]     = f2bf(xr[k]);
      a.u[2 * i + 1] = f2bf(xr[k + 1]);
    }
#pragma unroll
    for (int j = 0; j < 8; ++j) {
      int k = kc * 32 + ((j < 4) ? 0 : 16) + hi * 8 + (j & 3) * 2;
      b.u[2 * j]     = f2bf(W[(size_t)k * 16 + lanelo]);
      b.u[2 * j + 1] = f2bf(W[(size_t)(k + 1) * 16 + lanelo]);
    }
    c = __builtin_amdgcn_wmma_f32_16x16x32_bf16(false, a.v, false, b.v, (short)0, c,
                                                false, false);
  }
#pragma unroll
  for (int r = 0; r < 8; ++r)
    hout[(size_t)(rowBase + r + hi * 8) * 16 + lanelo] = c[r];
}

// ---------------- attention coefficients per node --------------------------------
__global__ void att_k(const float* __restrict__ h, const float* __restrict__ atts,
                      const float* __restrict__ attd, float* __restrict__ as_,
                      float* __restrict__ ad_, int n, int H) {
  int i = blockIdx.x * blockDim.x + threadIdx.x;
  if (i >= n) return;
  for (int hd = 0; hd < H; ++hd) {
    const float* hr = h + (size_t)i * H * HIDC + hd * HIDC;
    float ss = 0.f, sd = 0.f;
#pragma unroll
    for (int c = 0; c < HIDC; ++c) {
      float v = hr[c];
      ss += v * atts[hd * HIDC + c];
      sd += v * attd[hd * HIDC + c];
    }
    as_[(size_t)i * H + hd] = ss;
    ad_[(size_t)i * H + hd] = sd;
  }
}

// ---------------- segment softmax: pass A (max) ----------------------------------
__global__ void edge_max_k(const int* __restrict__ src, const int* __restrict__ dst,
                           const float* __restrict__ as_, const float* __restrict__ ad_,
                           unsigned* __restrict__ m, unsigned total, int E, int H) {
  unsigned i = blockIdx.x * blockDim.x + threadIdx.x;
  if (i >= total) return;
  unsigned eid = i / (unsigned)H; int hd = (int)(i % (unsigned)H);
  int s, d;
  if ((int)eid < E) { s = src[eid]; d = dst[eid]; } else { s = d = (int)eid - E; }
  float e = lrelu(as_[(size_t)s * H + hd] + ad_[(size_t)d * H + hd]);
  atomicMax(&m[(size_t)d * H + hd], encOrd(e));
}

// ---------------- pass B (sum of exp) --------------------------------------------
__global__ void edge_sum_k(const int* __restrict__ src, const int* __restrict__ dst,
                           const float* __restrict__ as_, const float* __restrict__ ad_,
                           const unsigned* __restrict__ m, float* __restrict__ den,
                           unsigned total, int E, int H) {
  unsigned i = blockIdx.x * blockDim.x + threadIdx.x;
  if (i >= total) return;
  unsigned eid = i / (unsigned)H; int hd = (int)(i % (unsigned)H);
  int s, d;
  if ((int)eid < E) { s = src[eid]; d = dst[eid]; } else { s = d = (int)eid - E; }
  float e = lrelu(as_[(size_t)s * H + hd] + ad_[(size_t)d * H + hd]);
  float ex = __expf(e - decOrd(m[(size_t)d * H + hd]));
  atomicAdd(&den[(size_t)d * H + hd], ex);
}

// ---------------- pass C (weighted scatter-add), 1 thread per (edge,channel) -----
__global__ void edge_acc_k(const int* __restrict__ src, const int* __restrict__ dst,
                           const float* __restrict__ as_, const float* __restrict__ ad_,
                           const unsigned* __restrict__ m, const float* __restrict__ den,
                           const float* __restrict__ hv, float* __restrict__ out,
                           unsigned total, int E, int H) {
  unsigned i = blockIdx.x * blockDim.x + threadIdx.x;
  if (i >= total) return;
  int HC = H * HIDC;
  unsigned eid = i / (unsigned)HC; int c = (int)(i % (unsigned)HC); int hd = c / HIDC;
  int s, d;
  if ((int)eid < E) { s = src[eid]; d = dst[eid]; } else { s = d = (int)eid - E; }
  float e = lrelu(as_[(size_t)s * H + hd] + ad_[(size_t)d * H + hd]);
  float ex = __expf(e - decOrd(m[(size_t)d * H + hd]));
  float alpha = ex / (den[(size_t)d * H + hd] + 1e-16f);
  atomicAdd(&out[(size_t)d * HC + c], hv[(size_t)s * HC + c] * alpha);
}

// ---------------- bias + ELU between layers --------------------------------------
__global__ void elu_bias_k(const float* __restrict__ in, const float* __restrict__ b,
                           float* __restrict__ out, unsigned tot) {
  unsigned i = blockIdx.x * blockDim.x + threadIdx.x;
  if (i >= tot) return;
  float v = in[i] + b[i & 63u];
  out[i] = v > 0.f ? v : (__expf(v) - 1.f);
}

// ---------------- fused edge-MLP + fc head (weights staged in LDS) ---------------
__global__ void final_k(const int* __restrict__ src, const int* __restrict__ dst,
                        const float* __restrict__ ea, const float* __restrict__ h2,
                        const float* __restrict__ b2, const float* __restrict__ mW1,
                        const float* __restrict__ mb1, const float* __restrict__ mW2,
                        const float* __restrict__ mb2, const float* __restrict__ fcW,
                        const float* __restrict__ fcb, float* __restrict__ out, int E) {
  __shared__ float sW1[8 * 16], sW2[16 * 16], sb1[16], sb2[16], sfc[49];
  for (int t = threadIdx.x; t < 128; t += blockDim.x) sW1[t] = mW1[t];
  for (int t = threadIdx.x; t < 256; t += blockDim.x) sW2[t] = mW2[t];
  for (int t = threadIdx.x; t < 16;  t += blockDim.x) { sb1[t] = mb1[t]; sb2[t] = mb2[t]; }
  for (int t = threadIdx.x; t < 48;  t += blockDim.x) sfc[t] = fcW[t];
  if (threadIdx.x == 0) sfc[48] = fcb[0];
  __syncthreads();
  int e = blockIdx.x * blockDim.x + threadIdx.x;
  if (e >= E) return;
  const float* er = ea + (size_t)e * 16;
  float ef[8];
#pragma unroll
  for (int j = 0; j < 8; ++j) ef[j] = er[j] * er[8 + j];
  float t1[16];
#pragma unroll
  for (int c = 0; c < 16; ++c) {
    float v = sb1[c];
#pragma unroll
    for (int j = 0; j < 8; ++j) v += ef[j] * sW1[j * 16 + c];
    t1[c] = v > 0.f ? v : 0.f;
  }
  float acc = sfc[48];
#pragma unroll
  for (int c = 0; c < 16; ++c) {
    float v = sb2[c];
#pragma unroll
    for (int k = 0; k < 16; ++k) v += t1[k] * sW2[k * 16 + c];
    acc += v * sfc[32 + c];
  }
  int s = src[e], d = dst[e];
  const float* hs = h2 + (size_t)s * 16;
  const float* hd = h2 + (size_t)d * 16;
#pragma unroll
  for (int c = 0; c < 16; ++c) {
    float bb = b2[c];
    acc += (hs[c] + bb) * sfc[c] + (hd[c] + bb) * sfc[16 + c];
  }
  out[e] = acc;
}

extern "C" void kernel_launch(void* const* d_in, const int* in_sizes, int n_in,
                              void* d_out, int out_size, void* d_ws, size_t ws_size,
                              hipStream_t stream) {
  (void)n_in; (void)out_size; (void)ws_size;
  const float* x    = (const float*)d_in[0];
  const int*   ei   = (const int*)d_in[1];
  const float* ea   = (const float*)d_in[2];
  const float* W1   = (const float*)d_in[5];
  const float* as1w = (const float*)d_in[6];
  const float* ad1w = (const float*)d_in[7];
  const float* b1   = (const float*)d_in[8];
  const float* W2   = (const float*)d_in[9];
  const float* as2w = (const float*)d_in[10];
  const float* ad2w = (const float*)d_in[11];
  const float* b2   = (const float*)d_in[12];
  const float* mW1  = (const float*)d_in[13];
  const float* mb1  = (const float*)d_in[14];
  const float* mW2  = (const float*)d_in[15];
  const float* mb2  = (const float*)d_in[16];
  const float* fcW  = (const float*)d_in[17];
  const float* fcb  = (const float*)d_in[18];
  float* out = (float*)d_out;

  const int N  = in_sizes[0] / 32;   // F_IN = 32
  const int E  = in_sizes[1] / 2;
  const int Ep = E + N;              // edges + self loops
  const int* srcI = ei;
  const int* dstI = ei + E;

  float* F = (float*)d_ws;           // 180*N floats total (72 MB @ N=100k)
  float*    h1   = F;
  float*    as1  = F + (size_t)N * 64;
  float*    ad1  = F + (size_t)N * 68;
  unsigned* m1   = (unsigned*)(F + (size_t)N * 72);
  float*    den1 = F + (size_t)N * 76;
  float*    out1 = F + (size_t)N * 80;
  float*    h2   = F + (size_t)N * 144;
  float*    as2  = F + (size_t)N * 160;
  float*    ad2  = F + (size_t)N * 161;
  unsigned* m2   = (unsigned*)(F + (size_t)N * 162);
  float*    den2 = F + (size_t)N * 163;
  float*    out2 = F + (size_t)N * 164;

  const int B = 256;
  auto g = [](long long t, int b) { return (unsigned)((t + b - 1) / b); };
  const int waves = (N + 15) / 16;
  const int gblk  = (waves + 7) / 8;   // 8 waves / 256-thread block

  // ---- GAT layer 1 (4 heads x 16) ----
  init1_k<<<g((long long)N * 64, B), B, 0, stream>>>(out1, den1, m1, N);
  gemm1_wmma_k<<<gblk, B, 0, stream>>>(x, W1, h1, N);
  att_k<<<g(N, B), B, 0, stream>>>(h1, as1w, ad1w, as1, ad1, N, 4);
  edge_max_k<<<g((long long)Ep * 4, B), B, 0, stream>>>(srcI, dstI, as1, ad1, m1,
                                                        (unsigned)((long long)Ep * 4), E, 4);
  edge_sum_k<<<g((long long)Ep * 4, B), B, 0, stream>>>(srcI, dstI, as1, ad1, m1, den1,
                                                        (unsigned)((long long)Ep * 4), E, 4);
  edge_acc_k<<<g((long long)Ep * 64, B), B, 0, stream>>>(srcI, dstI, as1, ad1, m1, den1,
                                                         h1, out1,
                                                         (unsigned)((long long)Ep * 64), E, 4);
  elu_bias_k<<<g((long long)N * 64, B), B, 0, stream>>>(out1, b1, h1, (unsigned)N * 64u);

  // ---- GAT layer 2 (1 head x 16) ----
  init2_k<<<g((long long)N * 16, B), B, 0, stream>>>(out2, den2, m2, N);
  gemm2_wmma_k<<<gblk, B, 0, stream>>>(h1, W2, h2, N);
  att_k<<<g(N, B), B, 0, stream>>>(h2, as2w, ad2w, as2, ad2, N, 1);
  edge_max_k<<<g(Ep, B), B, 0, stream>>>(srcI, dstI, as2, ad2, m2, (unsigned)Ep, E, 1);
  edge_sum_k<<<g(Ep, B), B, 0, stream>>>(srcI, dstI, as2, ad2, m2, den2, (unsigned)Ep, E, 1);
  edge_acc_k<<<g((long long)Ep * 16, B), B, 0, stream>>>(srcI, dstI, as2, ad2, m2, den2,
                                                         h2, out2,
                                                         (unsigned)((long long)Ep * 16), E, 1);

  // ---- fused edge MLP + fc head ----
  final_k<<<g(E, B), B, 0, stream>>>(srcI, dstI, ea, out2, b2, mW1, mb1, mW2, mb2,
                                     fcW, fcb, out, E);
}
